// GraphGenerator_43568148250689
// MI455X (gfx1250) — matrix operand order repaired and checked by reading, not compiled, not run on hardware
//
#include <hip/hip_runtime.h>
#include <hip/hip_bf16.h>
#include <math.h>

typedef __attribute__((ext_vector_type(16))) _Float16 v16h;
typedef __attribute__((ext_vector_type(8)))  _Float16 v8h;
typedef __attribute__((ext_vector_type(8)))  float    v8f;
typedef __attribute__((ext_vector_type(4)))  unsigned int v4u;
typedef __attribute__((ext_vector_type(8)))  int      v8i;
typedef __attribute__((ext_vector_type(4)))  int      v4i;

#define WAVE_LDS_FENCE() asm volatile("s_wait_dscnt 0x0" ::: "memory")

#if defined(__has_builtin)
#  if __has_builtin(__builtin_amdgcn_tensor_load_to_lds) && __has_builtin(__builtin_amdgcn_s_wait_tensorcnt)
#    define HAVE_TDM 1
#  endif
#endif
#ifndef HAVE_TDM
#  define HAVE_TDM 0
#endif

#if HAVE_TDM
// 1-D Tensor Data Mover copy: global -> LDS, `ndw` DWORDs (descriptor per ISA 08 §8).
// Issue from a single wave; EXEC is ignored by TENSOR_LOAD_TO_LDS.
__device__ inline void tdm_load_1d(void* lds_dst, const void* gsrc, unsigned int ndw) {
  unsigned int lds = (unsigned int)(uintptr_t)lds_dst;     // low 32 bits = LDS offset
  unsigned long long ga = (unsigned long long)(uintptr_t)gsrc;
  v4u g0;
  g0[0] = 1u;                                              // count=1, user descriptor
  g0[1] = lds;                                             // lds_addr
  g0[2] = (unsigned int)ga;                                // global_addr[31:0]
  g0[3] = (unsigned int)((ga >> 32) & 0x1FFFFFFu) | (2u << 30); // global_addr[56:32], type=2
  v8i g1;
  g1[0] = (int)(2u << 16);                                 // workgroup_mask=0, data_size=4B
  g1[1] = (int)((ndw & 0xFFFFu) << 16);                    // tensor_dim0[15:0]
  g1[2] = (int)(((ndw >> 16) & 0xFFFFu) | (1u << 16));     // tensor_dim0[31:16], tensor_dim1=1
  g1[3] = (int)((ndw & 0xFFFFu) << 16);                    // tensor_dim1 hi=0, tile_dim0=ndw
  g1[4] = 1;                                               // tile_dim1=1, tile_dim2=0
  g1[5] = (int)ndw;                                        // tensor_dim0_stride[31:0]
  g1[6] = 0;                                               // stride hi / dim1_stride lo
  g1[7] = 0;
  v4i z4 = {0, 0, 0, 0};
#  if __has_include(<hip/amd_detail/amd_gfx1250_TDM.h>)
  v8i z8 = {0, 0, 0, 0, 0, 0, 0, 0};
  __builtin_amdgcn_tensor_load_to_lds(g0, g1, z4, z4, z8, 0);
#  else
  __builtin_amdgcn_tensor_load_to_lds(g0, g1, z4, z4, 0);
#  endif
  __builtin_amdgcn_s_wait_tensorcnt(0);
}
#endif

__device__ inline v8f wmma_f16(v16h a, v16h b, v8f c) {
  // D = A(16x32 f16) * B(32x16 f16) + C(16x16 f32)
  return __builtin_amdgcn_wmma_f32_16x16x32_f16(false, a, false, b, (short)0, c, false, false);
}

// Build a 16x32 f16 A-fragment from an LDS tile (row-major, `stride` halves/row),
// per CDNA5 ISA 16-bit A layout: lanes 0-15 -> K=kbase+[0..7]/[16..23],
// lanes 16-31 -> K=kbase+[8..15]/[24..31].
__device__ inline v16h load_afrag(const _Float16* base, int stride, int kbase, int lane) {
  int m = lane & 15, hi = lane >> 4;
  const _Float16* p = base + m * stride + kbase + hi * 8;
  v8h lo = *(const v8h*)p;
  v8h hh = *(const v8h*)(p + 16);
  return __builtin_shufflevector(lo, hh, 0,1,2,3,4,5,6,7,8,9,10,11,12,13,14,15);
}

// Branch-free tanh-gelu via the identity 0.5*(1+tanh(z)) == sigmoid(2z):
// gelu(x) ~= x * sigmoid(1.5957691*(x + 0.044715 x^3)).
// One v_exp_f32 + one v_rcp_f32 per element, no divergence, co-executes with WMMA.
__device__ inline float gelu_fast(float x) {
  float x3 = x * x * x;
  float g2 = 1.5957691216057308f * fmaf(0.044715f, x3, x);
  float u = __expf(-g2);
  return x * __builtin_amdgcn_rcpf(1.f + u);
}

// Scatter D fragment (16x16 f32) + bias -> gelu -> f16 LDS stage tile (stride 64).
__device__ inline void store_gelu(_Float16 (*st)[64], v8f c, int nt, int lane, const float* bias) {
  int n = lane & 15, mb = (lane >> 4) * 8;
  int col = nt * 16 + n;
  float bb = bias[col];           // padded bias is 0 for col>=50, keeps padding at gelu(0)=0
  #pragma unroll
  for (int r = 0; r < 8; ++r) {
    float xv = c[r] + bb;
    st[mb + r][col] = (_Float16)gelu_fast(xv);
  }
}

__device__ inline void atomicMaxF(float* p, float v) {
#if defined(__has_builtin) && __has_builtin(__hip_atomic_fetch_max)
  __hip_atomic_fetch_max(p, v, __ATOMIC_RELAXED, __HIP_MEMORY_SCOPE_AGENT);
#else
  int* ip = (int*)p;
  int old = __float_as_int(*p);
  while (__int_as_float(old) < v) {
    int assumed = old;
    old = atomicCAS(ip, assumed, __float_as_int(v));
    if (old == assumed) break;
  }
#endif
}

// ---------------------------------------------------------------- constants
#define HID 25
#define HEADS 5
#define CH 5
#define MLPH 50
// packed MLP weight region (halves): W0 4x[32x16] | W1 4x2x[32x16] | W2 | W3 2x[32x16]
#define PW0 0
#define PW1 2048
#define PW2 6144
#define PW3 10240
#define PMLP_TOT 11264
#define PQKVS_TOT 12288   // 3 layers * 4 mats * 2 ntiles * 32 * 16
#define EW 8              // waves per block
#define TPW_E 16          // edge tiles per wave
#define TPW_N 8           // node tiles per wave

// ---------------------------------------------------------------- weight packing
__global__ __launch_bounds__(256) void pack_weights(
    const float* __restrict__ W0, const float* __restrict__ W1,
    const float* __restrict__ W2, const float* __restrict__ W3,
    const float* __restrict__ Wq, const float* __restrict__ Wk,
    const float* __restrict__ Wv, const float* __restrict__ Ws,
    _Float16* __restrict__ pmlp, _Float16* __restrict__ pqkvs)
{
  int idx = blockIdx.x * blockDim.x + threadIdx.x;
  if (idx < PMLP_TOT) {
    float val = 0.f;
    if (idx < PW1) {                       // W0: 13x50 -> 4 ntiles, K=32
      int nt = idx >> 9, r = idx & 511, kk = (((r & 15) >> 3) << 4) + ((r >> 8) & 1) * 8 + (r & 7), n = (r >> 4) & 15, col = nt * 16 + n;
      if (kk < 13 && col < MLPH) val = W0[kk * MLPH + col];
    } else if (idx < PW2) {                // W1: 50x50 -> 4 ntiles x 2 K-chunks
      int j = idx - PW1;
      int nt = j >> 10, r = j & 1023, ch = r >> 9, k2 = (((r & 15) >> 3) << 4) + ((r >> 8) & 1) * 8 + (r & 7), n = (r >> 4) & 15;
      int srck = ch * 32 + k2, col = nt * 16 + n;
      if (srck < MLPH && col < MLPH) val = W1[srck * MLPH + col];
    } else if (idx < PW3) {                // W2
      int j = idx - PW2;
      int nt = j >> 10, r = j & 1023, ch = r >> 9, k2 = (((r & 15) >> 3) << 4) + ((r >> 8) & 1) * 8 + (r & 7), n = (r >> 4) & 15;
      int srck = ch * 32 + k2, col = nt * 16 + n;
      if (srck < MLPH && col < MLPH) val = W2[srck * MLPH + col];
    } else {                               // W3: 50x2 -> 1 ntile x 2 K-chunks
      int j = idx - PW3;
      int ch = j >> 9, k2 = (((j & 15) >> 3) << 4) + ((j >> 8) & 1) * 8 + (j & 7), n = (j >> 4) & 15;
      int srck = ch * 32 + k2;
      if (srck < MLPH && n < 2) val = W3[srck * 2 + n];
    }
    pmlp[idx] = (_Float16)val;
  } else if (idx < PMLP_TOT + PQKVS_TOT) { // q/k/v/skip: 25x25 -> 2 ntiles, K=32
    int j = idx - PMLP_TOT;
    int l = j / 4096, r = j & 4095, mat = r >> 10, nt = (r >> 9) & 1, k2 = (((r & 15) >> 3) << 4) + ((r >> 8) & 1) * 8 + (r & 7), n = (r >> 4) & 15;
    int col = nt * 16 + n;
    float val = 0.f;
    const float* Wm = (mat == 0) ? Wq : (mat == 1) ? Wk : (mat == 2) ? Wv : Ws;
    if (k2 < HID && col < HID) val = Wm[l * HID * HID + k2 * HID + col];
    pqkvs[j] = (_Float16)val;
  }
}

// ---------------------------------------------------------------- node prep
__global__ __launch_bounds__(256) void prep_nodes(
    const float* __restrict__ x, const float* __restrict__ token,
    const float* __restrict__ ilW, const float* __restrict__ ilb,
    _Float16* __restrict__ xth, float* __restrict__ hn, int N)
{
  int n = blockIdx.x * blockDim.x + threadIdx.x;
  if (n >= N) return;
  float xv[6];
  xv[0] = x[n];
  #pragma unroll
  for (int i = 0; i < 5; ++i) xv[1 + i] = token[(size_t)n * 5 + i];
  #pragma unroll
  for (int i = 0; i < 6; ++i) xth[(size_t)n * 8 + i] = (_Float16)xv[i];
  xth[(size_t)n * 8 + 6] = (_Float16)0.f;
  xth[(size_t)n * 8 + 7] = (_Float16)0.f;
  #pragma unroll
  for (int j = 0; j < 32; ++j) {
    float acc = 0.f;
    if (j < HID) {
      acc = ilb[j];
      #pragma unroll
      for (int i = 0; i < 6; ++i) acc = fmaf(xv[i], ilW[i * HID + j], acc);
    }
    hn[(size_t)n * 32 + j] = acc;
  }
}

// ---------------------------------------------------------------- WMMA edge MLP + gumbel hard argmax
__global__ __launch_bounds__(256) void edge_mlp(
    const int* __restrict__ ei, const float* __restrict__ ea,
    const float* __restrict__ gum, const _Float16* __restrict__ xth,
    const _Float16* __restrict__ pmlp,
    const float* __restrict__ b0, const float* __restrict__ b1,
    const float* __restrict__ b2, const float* __restrict__ b3,
    float* __restrict__ evout, int E)
{
  __shared__ __align__(32) _Float16 wl[PMLP_TOT];
  __shared__ __align__(16) _Float16 stage0[EW][16][32]; // per-wave layer-0 input (13 live cols)
  __shared__ __align__(16) _Float16 stage[EW][16][64];  // per-wave activation tile
  __shared__ float lbuf[EW][16][2];
  __shared__ float bl[192];                             // biases b0|b1|b2 padded to 64
  int t = threadIdx.x;
#if HAVE_TDM
  if (t < 32) tdm_load_1d(wl, pmlp, PMLP_TOT / 2);      // 22.5 KB via Tensor Data Mover
#else
  {
    const uint4* s = (const uint4*)pmlp;
    uint4* d = (uint4*)wl;
    for (int i = t; i < PMLP_TOT / 8; i += 256) d[i] = s[i];
  }
#endif
  for (int i = t; i < 192; i += 256) {
    int g = i & 63, sel = i >> 6;
    const float* bp = (sel == 0) ? b0 : (sel == 1) ? b1 : b2;
    bl[i] = (g < MLPH) ? bp[g] : 0.f;
  }
  __syncthreads();

  int wave = t >> 5, lane = t & 31, li = lane & 15, hi = lane >> 4;
  _Float16(*s0)[32] = stage0[wave];
  _Float16(*st)[64] = stage[wave];
  // zero the K-pad of the layer-0 input once (cols 13..31 never rewritten)
  {
    int c0 = 13 + hi * 10;                // lanes 0-15 cols 13..22, lanes 16-31 cols 23..31
    int c1 = (hi == 0) ? 23 : 32;
    for (int j = c0; j < c1; ++j) s0[li][j] = (_Float16)0.f;
  }
  int tile0 = (blockIdx.x * EW + wave) * TPW_E;
  for (int tt = 0; tt < TPW_E; ++tt) {
    int tile = tile0 + tt;
    if (tile * 16 >= E) break;
    int e = tile * 16 + li;
    if (tt + 1 < TPW_E && (tile + 1) * 16 < E) {        // global_prefetch next tile
      __builtin_prefetch(ei + e + 16, 0, 1);
      __builtin_prefetch(gum + (size_t)(e + 16) * 2, 0, 1);
    }
    // gather: lanes 0-15 -> src features + edge_attr, lanes 16-31 -> dst features
    if (hi == 0) {
      int sn = ei[e];
      v8h f = *(const v8h*)(xth + (size_t)sn * 8);
      #pragma unroll
      for (int j = 0; j < 6; ++j) s0[li][j] = f[j];
      s0[li][12] = (_Float16)ea[e];
    } else {
      int dn = ei[E + e];
      v8h f = *(const v8h*)(xth + (size_t)dn * 8);
      #pragma unroll
      for (int j = 0; j < 6; ++j) s0[li][6 + j] = f[j];
    }
    WAVE_LDS_FENCE();
    // ---- layer 0: K=32 (13 valid), N=64 (50 valid)
    {
      v16h a = load_afrag(&s0[0][0], 32, 0, lane);
      #pragma unroll
      for (int nt = 0; nt < 4; ++nt) {
        v16h b = *(const v16h*)(wl + PW0 + nt * 512 + lane * 16);
        v8f c = {};
        c = wmma_f16(a, b, c);
        store_gelu(st, c, nt, lane, bl);
      }
    }
    WAVE_LDS_FENCE();
    // ---- layer 1: K=64, N=64
    {
      v16h a0 = load_afrag(&st[0][0], 64, 0, lane);
      v16h a1 = load_afrag(&st[0][0], 64, 32, lane);
      #pragma unroll
      for (int nt = 0; nt < 4; ++nt) {
        v16h bA = *(const v16h*)(wl + PW1 + nt * 1024 + lane * 16);
        v16h bB = *(const v16h*)(wl + PW1 + nt * 1024 + 512 + lane * 16);
        v8f c = {};
        c = wmma_f16(a0, bA, c);
        c = wmma_f16(a1, bB, c);
        store_gelu(st, c, nt, lane, bl + 64);
      }
    }
    WAVE_LDS_FENCE();
    // ---- layer 2: K=64, N=64
    {
      v16h a0 = load_afrag(&st[0][0], 64, 0, lane);
      v16h a1 = load_afrag(&st[0][0], 64, 32, lane);
      #pragma unroll
      for (int nt = 0; nt < 4; ++nt) {
        v16h bA = *(const v16h*)(wl + PW2 + nt * 1024 + lane * 16);
        v16h bB = *(const v16h*)(wl + PW2 + nt * 1024 + 512 + lane * 16);
        v8f c = {};
        c = wmma_f16(a0, bA, c);
        c = wmma_f16(a1, bB, c);
        store_gelu(st, c, nt, lane, bl + 128);
      }
    }
    WAVE_LDS_FENCE();
    // ---- layer 3: K=64, N=16 (2 valid) -> logits -> gumbel hard argmax
    {
      v16h a0 = load_afrag(&st[0][0], 64, 0, lane);
      v16h a1 = load_afrag(&st[0][0], 64, 32, lane);
      v16h bA = *(const v16h*)(wl + PW3 + lane * 16);
      v16h bB = *(const v16h*)(wl + PW3 + 512 + lane * 16);
      v8f c = {};
      c = wmma_f16(a0, bA, c);
      c = wmma_f16(a1, bB, c);
      if (li < 2) {
        float bb = b3[li];
        #pragma unroll
        for (int r = 0; r < 8; ++r) lbuf[wave][hi * 8 + r][li] = c[r] + bb;
      }
      WAVE_LDS_FENCE();
      if (hi == 0) {
        float l0 = lbuf[wave][li][0] + gum[(size_t)e * 2 + 0];
        float l1 = lbuf[wave][li][1] + gum[(size_t)e * 2 + 1];
        evout[e] = (l1 > l0) ? 1.f : 0.f;   // argmax picks index 0 on ties
      }
    }
  }
}

// ---------------------------------------------------------------- WMMA q/k/v/skip projections (one layer)
__global__ __launch_bounds__(256) void node_qkvs(
    const float* __restrict__ hn, const _Float16* __restrict__ pw,
    const float* __restrict__ bq, const float* __restrict__ bk,
    const float* __restrict__ bv, const float* __restrict__ bs,
    float* __restrict__ q, float* __restrict__ k, float* __restrict__ v,
    float* __restrict__ hs, int N)
{
  __shared__ __align__(32) _Float16 wl[4096];
  __shared__ __align__(16) _Float16 hstage[EW][16][32];
  __shared__ float bl[4][32];
  int t = threadIdx.x;
#if HAVE_TDM
  if (t < 32) tdm_load_1d(wl, pw, 2048);               // 8 KB via Tensor Data Mover
#else
  {
    const uint4* s = (const uint4*)pw;
    uint4* d = (uint4*)wl;
    for (int i = t; i < 512; i += 256) d[i] = s[i];
  }
#endif
  if (t < 128) {
    int mat = t >> 5, col = t & 31;
    const float* bp = (mat == 0) ? bq : (mat == 1) ? bk : (mat == 2) ? bv : bs;
    bl[mat][col] = (col < HID) ? bp[col] : 0.f;
  }
  __syncthreads();

  int wave = t >> 5, lane = t & 31, li = lane & 15, hi = lane >> 4;
  int tile0 = (blockIdx.x * EW + wave) * TPW_N;
  for (int tt = 0; tt < TPW_N; ++tt) {
    int tile = tile0 + tt;
    if (tile * 16 >= N) break;
    int node = tile * 16 + li;
    {
      const float* hp = hn + (size_t)node * 32;
      int c0 = hi * 16;
      #pragma unroll
      for (int j = 0; j < 16; ++j) {
        int col = c0 + j;
        hstage[wave][li][col] = (_Float16)((col < HID) ? hp[col] : 0.f);
      }
    }
    WAVE_LDS_FENCE();
    v16h a = load_afrag(&hstage[wave][0][0], 32, 0, lane);
    #pragma unroll
    for (int mat = 0; mat < 4; ++mat) {
      float* op = (mat == 0) ? q : (mat == 1) ? k : (mat == 2) ? v : hs;
      #pragma unroll
      for (int nt = 0; nt < 2; ++nt) {
        v16h b = *(const v16h*)(wl + mat * 1024 + nt * 512 + lane * 16);
        v8f c = {};
        c = wmma_f16(a, b, c);
        int n = lane & 15, col = nt * 16 + n, mb = hi * 8;
        float bb = bl[mat][col];
        #pragma unroll
        for (int r = 0; r < 8; ++r)
          op[(size_t)(tile * 16 + mb + r) * 32 + col] = c[r] + bb;
      }
    }
  }
}

// ---------------------------------------------------------------- attention scatter passes
__global__ __launch_bounds__(256) void init_layer(
    float* __restrict__ agg, float* __restrict__ amax, float* __restrict__ den, int N)
{
  int i = blockIdx.x * blockDim.x + threadIdx.x;
  if (i < N * 32) agg[i] = 0.f;
  if (i < N * 8) { amax[i] = -3.4e38f; den[i] = 0.f; }
}

__device__ inline void edge_alpha(int e, int E, const int* ei, const float* ea,
                                  const float* ev, const float* q, const float* k,
                                  const float* We, const float* be,
                                  int& sn, int& dn, float* evec, float* alpha)
{
  sn = ei[e];
  dn = ei[E + e];
  float a_ = ea[e], v_ = ev[e];
  #pragma unroll
  for (int h = 0; h < HEADS; ++h) alpha[h] = 0.f;
  #pragma unroll
  for (int j = 0; j < HID; ++j) {
    float ej = fmaf(a_, We[j], fmaf(v_, We[HID + j], be[j]));
    evec[j] = ej;
    float kj = k[(size_t)sn * 32 + j] + ej;
    alpha[j / CH] = fmaf(q[(size_t)dn * 32 + j], kj, alpha[j / CH]);
  }
  #pragma unroll
  for (int h = 0; h < HEADS; ++h) alpha[h] *= 0.44721359549995794f; // 1/sqrt(C)
}

__global__ __launch_bounds__(256) void attn_max(
    const int* __restrict__ ei, const float* __restrict__ ea, const float* __restrict__ ev,
    const float* __restrict__ q, const float* __restrict__ k,
    const float* __restrict__ We, const float* __restrict__ be,
    float* __restrict__ amax, int E)
{
  int e = blockIdx.x * blockDim.x + threadIdx.x;
  if (e >= E) return;
  int sn, dn; float evec[HID], alpha[HEADS];
  edge_alpha(e, E, ei, ea, ev, q, k, We, be, sn, dn, evec, alpha);
  #pragma unroll
  for (int h = 0; h < HEADS; ++h) atomicMaxF(&amax[(size_t)dn * 8 + h], alpha[h]);
}

__global__ __launch_bounds__(256) void attn_sum(
    const int* __restrict__ ei, const float* __restrict__ ea, const float* __restrict__ ev,
    const float* __restrict__ q, const float* __restrict__ k,
    const float* __restrict__ We, const float* __restrict__ be,
    const float* __restrict__ amax, float* __restrict__ den, int E)
{
  int e = blockIdx.x * blockDim.x + threadIdx.x;
  if (e >= E) return;
  int sn, dn; float evec[HID], alpha[HEADS];
  edge_alpha(e, E, ei, ea, ev, q, k, We, be, sn, dn, evec, alpha);
  #pragma unroll
  for (int h = 0; h < HEADS; ++h) {
    float ex = __expf(alpha[h] - amax[(size_t)dn * 8 + h]);
    atomicAdd(&den[(size_t)dn * 8 + h], ex);
  }
}

__global__ __launch_bounds__(256) void attn_agg(
    const int* __restrict__ ei, const float* __restrict__ ea, const float* __restrict__ ev,
    const float* __restrict__ q, const float* __restrict__ k, const float* __restrict__ v,
    const float* __restrict__ We, const float* __restrict__ be,
    const float* __restrict__ amax, const float* __restrict__ den,
    float* __restrict__ agg, int E)
{
  int e = blockIdx.x * blockDim.x + threadIdx.x;
  if (e >= E) return;
  int sn, dn; float evec[HID], alpha[HEADS];
  edge_alpha(e, E, ei, ea, ev, q, k, We, be, sn, dn, evec, alpha);
  float w[HEADS];
  #pragma unroll
  for (int h = 0; h < HEADS; ++h) {
    float ex = __expf(alpha[h] - amax[(size_t)dn * 8 + h]);
    w[h] = ex * __builtin_amdgcn_rcpf(den[(size_t)dn * 8 + h] + 1e-16f);
  }
  #pragma unroll
  for (int j = 0; j < HID; ++j) {
    float msg = (v[(size_t)sn * 32 + j] + evec[j]) * w[j / CH];
    atomicAdd(&agg[(size_t)dn * 32 + j], msg);
  }
}

__global__ __launch_bounds__(256) void node_update(
    float* __restrict__ hn, const float* __restrict__ agg,
    const float* __restrict__ hs, int N)
{
  int i = blockIdx.x * blockDim.x + threadIdx.x;
  if (i < N * 32) hn[i] = agg[i] + hs[i];
}

__global__ __launch_bounds__(256) void final_out(
    const float* __restrict__ hn, const float* __restrict__ oW,
    const float* __restrict__ ob, float* __restrict__ out, int N)
{
  int n = blockIdx.x * blockDim.x + threadIdx.x;
  if (n >= N) return;
  float acc = ob[0];
  #pragma unroll
  for (int j = 0; j < HID; ++j) acc = fmaf(hn[(size_t)n * 32 + j], oW[j], acc);
  out[n] = __builtin_amdgcn_rcpf(1.f + __expf(-acc));
}

// ---------------------------------------------------------------- host
extern "C" void kernel_launch(void* const* d_in, const int* in_sizes, int n_in,
                              void* d_out, int out_size, void* d_ws, size_t ws_size,
                              hipStream_t stream)
{
  const float* x        = (const float*)d_in[0];
  const float* token    = (const float*)d_in[1];
  const float* edge_attr= (const float*)d_in[2];
  const float* gumbel   = (const float*)d_in[3];
  const int*   ei       = (const int*)d_in[4];
  const float* W0 = (const float*)d_in[5];  const float* b0 = (const float*)d_in[6];
  const float* W1 = (const float*)d_in[7];  const float* b1 = (const float*)d_in[8];
  const float* W2 = (const float*)d_in[9];  const float* b2 = (const float*)d_in[10];
  const float* W3 = (const float*)d_in[11]; const float* b3 = (const float*)d_in[12];
  const float* ilW = (const float*)d_in[13]; const float* ilb = (const float*)d_in[14];
  const float* Wq = (const float*)d_in[15]; const float* bq = (const float*)d_in[16];
  const float* Wk = (const float*)d_in[17]; const float* bk = (const float*)d_in[18];
  const float* Wv = (const float*)d_in[19]; const float* bv = (const float*)d_in[20];
  const float* We = (const float*)d_in[21]; const float* be = (const float*)d_in[22];
  const float* Wsk= (const float*)d_in[23]; const float* bsk= (const float*)d_in[24];
  const float* oW = (const float*)d_in[25]; const float* ob = (const float*)d_in[26];

  const int N = in_sizes[0];   // x is (N,1)
  const int E = in_sizes[2];   // edge_attr is (E,1)

  char* ws = (char*)d_ws;
  size_t off = 0;
  auto alloc = [&](size_t bytes) -> void* {
    void* p = ws + off;
    off = (off + bytes + 255) & ~(size_t)255;
    return p;
  };
  _Float16* xth  = (_Float16*)alloc((size_t)N * 8 * 2);
  float* hn   = (float*)alloc((size_t)N * 32 * 4);
  float* q    = (float*)alloc((size_t)N * 32 * 4);
  float* k    = (float*)alloc((size_t)N * 32 * 4);
  float* v    = (float*)alloc((size_t)N * 32 * 4);
  float* hs   = (float*)alloc((size_t)N * 32 * 4);
  float* agg  = (float*)alloc((size_t)N * 32 * 4);
  float* amax = (float*)alloc((size_t)N * 8 * 4);
  float* den  = (float*)alloc((size_t)N * 8 * 4);
  float* evv  = (float*)alloc((size_t)E * 4);
  _Float16* pmlp  = (_Float16*)alloc(PMLP_TOT * 2);
  _Float16* pqkvs = (_Float16*)alloc(PQKVS_TOT * 2);
  (void)ws_size; (void)n_in; (void)out_size;

  pack_weights<<<(PMLP_TOT + PQKVS_TOT + 255) / 256, 256, 0, stream>>>(
      W0, W1, W2, W3, Wq, Wk, Wv, Wsk, pmlp, pqkvs);
  prep_nodes<<<(N + 255) / 256, 256, 0, stream>>>(x, token, ilW, ilb, xth, hn, N);

  int ntE = (E + 15) / 16;
  int gE = (ntE + EW * TPW_E - 1) / (EW * TPW_E);
  edge_mlp<<<gE, 256, 0, stream>>>(ei, edge_attr, gumbel, xth, pmlp,
                                   b0, b1, b2, b3, evv, E);

  int ntN = (N + 15) / 16;
  int gN = (ntN + EW * TPW_N - 1) / (EW * TPW_N);
  int gI = (N * 32 + 255) / 256;
  int gEdge = (E + 255) / 256;

  for (int l = 0; l < 3; ++l) {
    init_layer<<<gI, 256, 0, stream>>>(agg, amax, den, N);
    node_qkvs<<<gN, 256, 0, stream>>>(hn, pqkvs + l * 4096,
                                      bq + l * HID, bk + l * HID, bv + l * HID, bsk + l * HID,
                                      q, k, v, hs, N);
    attn_max<<<gEdge, 256, 0, stream>>>(ei, edge_attr, evv, q, k,
                                        We + l * 2 * HID, be + l * HID, amax, E);
    attn_sum<<<gEdge, 256, 0, stream>>>(ei, edge_attr, evv, q, k,
                                        We + l * 2 * HID, be + l * HID, amax, den, E);
    attn_agg<<<gEdge, 256, 0, stream>>>(ei, edge_attr, evv, q, k, v,
                                        We + l * 2 * HID, be + l * HID, amax, den, agg, E);
    node_update<<<gI, 256, 0, stream>>>(hn, agg, hs, N);
  }
  final_out<<<(N + 255) / 256, 256, 0, stream>>>(hn, oW, ob, (float*)d_out, N);
}
